// E_GraphSAGE_Entropy_53171695124549
// MI455X (gfx1250) — matrix operand order repaired and checked by reading, not compiled, not run on hardware
//
#include <hip/hip_runtime.h>
#include <hip/hip_bf16.h>

// ---------------------------------------------------------------------------
// E-GraphSAGE on MI455X (gfx1250).
// Pipeline: memset ws -> scatter1 (deg + layer1 msg sums) -> dense1 (h1)
//        -> scatter2 (h1[src] sums) -> dense2 (h2, stored f16)
//        -> fused WMMA edge classifier (160->64->32->1).
// The classifier carries ~20 GFLOP of the ~21 GFLOP total; it runs on
// v_wmma_f32_16x16x32_f16 with f32 accumulation, grid-stride over 16-edge
// tiles so the LDS-staged f16 weights are reused across many tiles.
// ---------------------------------------------------------------------------

typedef __attribute__((ext_vector_type(16))) _Float16 v16h;
typedef __attribute__((ext_vector_type(8)))  _Float16 half8;
typedef __attribute__((ext_vector_type(8)))  float    v8f;

#define HID   64
#define EDIM  32
#define K1    160   // 2*HID + EDIM
#define WPB   4     // waves per block in classifier

// ---------------- scatter kernels (atomics into L2-resident buffers) -------

// one thread per (edge, attr component): coalesced ea reads, wide atomic issue
__global__ void scatter1_kernel(const int* __restrict__ eidx,
                                const float* __restrict__ ns,
                                const float* __restrict__ ea,
                                float* __restrict__ deg,
                                float* __restrict__ agg1, int E) {
    long long t = (long long)blockIdx.x * blockDim.x + threadIdx.x;
    if (t >= (long long)E * EDIM) return;
    int e = (int)(t >> 5);
    int j = (int)(t & 31);
    int d = eidx[E + e];
    float* a = agg1 + (size_t)d * 34;
    atomicAdd(&a[2 + j], ea[(size_t)e * EDIM + j]);
    if (j == 0) {
        int s = eidx[e];
        atomicAdd(&deg[d], 1.0f);
        atomicAdd(&a[0], ns[(size_t)s * 2 + 0]);
        atomicAdd(&a[1], ns[(size_t)s * 2 + 1]);
    }
}

__global__ void scatter2_kernel(const int* __restrict__ eidx,
                                const float* __restrict__ h1,
                                float* __restrict__ agg2, int E) {
    long long t = (long long)blockIdx.x * blockDim.x + threadIdx.x;
    if (t >= (long long)E * HID) return;
    int e = (int)(t >> 6);
    int j = (int)(t & 63);
    int s = eidx[e];
    int d = eidx[E + e];
    atomicAdd(&agg2[(size_t)d * HID + j], h1[(size_t)s * HID + j]);
}

// ---------------- node-level dense layers (small FLOP, VALU) ---------------

__global__ void dense1_kernel(const float* __restrict__ ns,
                              const float* __restrict__ deg,
                              const float* __restrict__ agg1,
                              const float* __restrict__ W1,
                              const float* __restrict__ b1,
                              float* __restrict__ h1, int N) {
    int t = blockIdx.x * blockDim.x + threadIdx.x;
    if (t >= N * HID) return;
    int n = t >> 6, o = t & 63;
    float inv = 1.0f / fmaxf(deg[n], 1.0f);
    const float* w = W1 + (size_t)o * 36;
    float s = b1[o] + ns[(size_t)n * 2] * w[0] + ns[(size_t)n * 2 + 1] * w[1];
    const float* a = agg1 + (size_t)n * 34;
    float s2 = 0.0f;
#pragma unroll
    for (int k = 0; k < 34; ++k) s2 += a[k] * w[2 + k];
    s += inv * s2;
    h1[(size_t)n * HID + o] = fmaxf(s, 0.0f);
}

__global__ void dense2_kernel(const float* __restrict__ h1,
                              const float* __restrict__ deg,
                              const float* __restrict__ agg2,
                              const float* __restrict__ agg1,
                              const float* __restrict__ W2,
                              const float* __restrict__ b2,
                              _Float16* __restrict__ h2h, int N) {
    int t = blockIdx.x * blockDim.x + threadIdx.x;
    if (t >= N * HID) return;
    int n = t >> 6, o = t & 63;
    float inv = 1.0f / fmaxf(deg[n], 1.0f);
    const float* w = W2 + (size_t)o * K1;
    float s = b2[o];
    const float* hn = h1 + (size_t)n * HID;
#pragma unroll
    for (int k = 0; k < HID; ++k) s += hn[k] * w[k];
    float s2 = 0.0f;
    const float* a2 = agg2 + (size_t)n * HID;
#pragma unroll
    for (int k = 0; k < HID; ++k) s2 += a2[k] * w[HID + k];
    const float* a1 = agg1 + (size_t)n * 34;
#pragma unroll
    for (int j = 0; j < EDIM; ++j) s2 += a1[2 + j] * w[2 * HID + j];
    s += inv * s2;
    h2h[(size_t)n * HID + o] = (_Float16)fmaxf(s, 0.0f);
}

// ---------------- WMMA fragment builders (CDNA5 16x16x32 f16 layouts) ------

// A (M=16 x K=32): lane L holds row M=L&15; halfs 0..7 -> K=(L>=16?8:0)+j,
// halfs 8..15 -> K=16+(L>=16?8:0)+j.  Two contiguous 16B LDS reads.
__device__ __forceinline__ v16h frag_a(const _Float16* tileBase, int rowStride,
                                       int kbase, int lane) {
    int m  = lane & 15;
    int ko = kbase + ((lane & 16) ? 8 : 0);
    const _Float16* p = tileBase + m * rowStride + ko;
    half8 lo = *(const half8*)p;
    half8 hi = *(const half8*)(p + 16);
    v16h a;
#pragma unroll
    for (int j = 0; j < 8; ++j) { a[j] = lo[j]; a[8 + j] = hi[j]; }
    return a;
}

// B (K=32 x N=16) with B[k][n] = W[n][k] (row-major weight rows):
// lane L holds N=L&15; halfs 0..7 -> K=(L>=16?8:0)+j, halfs 8..15 -> K=16+(L>=16?8:0)+j.
__device__ __forceinline__ v16h frag_b(const _Float16* w, int rowStride,
                                       int nbase, int kbase, int lane) {
    int n  = nbase + (lane & 15);
    int ko = kbase + ((lane & 16) ? 8 : 0);
    const _Float16* p = w + n * rowStride + ko;
    half8 lo = *(const half8*)p;
    half8 hi = *(const half8*)(p + 16);
    v16h b;
#pragma unroll
    for (int j = 0; j < 8; ++j) { b[j] = lo[j]; b[8 + j] = hi[j]; }
    return b;
}

// ---------------- fused per-edge classifier (the WMMA kernel) --------------

__global__ __launch_bounds__(WPB * 32)
void classifier_kernel(const int* __restrict__ eidx,
                       const float* __restrict__ ea,
                       const _Float16* __restrict__ h2h,
                       const float* __restrict__ Wc1, const float* __restrict__ bc1,
                       const float* __restrict__ Wc2, const float* __restrict__ bc2,
                       const float* __restrict__ Wc3, const float* __restrict__ bc3,
                       float* __restrict__ out, int E, int numTiles) {
    __shared__ __align__(16) _Float16 s_wc1[HID * K1];       // 20 KB
    __shared__ __align__(16) _Float16 s_wc2[32 * HID];       //  4 KB
    __shared__ float  s_wc3[32];
    __shared__ float  s_bc1[HID];
    __shared__ float  s_bc2[32];
    __shared__ float  s_bc3[1];
    __shared__ __align__(16) _Float16 s_rep[WPB][16 * K1];   // 20 KB
    __shared__ __align__(16) _Float16 s_z1[WPB][16 * HID];   //  8 KB
    __shared__ __align__(16) float    s_z2[WPB][16 * 32];    //  8 KB

    const int tid = threadIdx.x;
    for (int i = tid; i < HID * K1; i += blockDim.x) s_wc1[i] = (_Float16)Wc1[i];
    for (int i = tid; i < 32 * HID; i += blockDim.x) s_wc2[i] = (_Float16)Wc2[i];
    if (tid < 32) { s_wc3[tid] = Wc3[tid]; s_bc2[tid] = bc2[tid]; }
    if (tid < HID) s_bc1[tid] = bc1[tid];
    if (tid == 0)  s_bc3[0] = bc3[0];
    __syncthreads();

    const int wave = tid >> 5;
    const int lane = tid & 31;
    const int tileStride = gridDim.x * WPB;
    _Float16* rep = s_rep[wave];
    _Float16* z1  = s_z1[wave];
    float*    z2  = s_z2[wave];
    const int* srcIdx = eidx;
    const int* dstIdx = eidx + E;
    const int mOff = (lane & 16) ? 8 : 0;

    for (int tile = blockIdx.x * WPB + wave; tile < numTiles; tile += tileStride) {
        const int base = tile * 16;

        // prefetch next tile's edge_attr stream (global_prefetch_b8)
        {
            long long nbase = (long long)(tile + tileStride) * 16;
            if (nbase < E)
                __builtin_prefetch(ea + nbase * EDIM + lane * 16, 0, 1);
        }

        // ---- stage rep = [h2[src] | h2[dst] | f16(edge_attr)] : 16 x 160 f16
#pragma unroll
        for (int it = 0; it < 10; ++it) {
            int c  = it * 32 + lane;          // 320 chunks of 8 halfs
            int el = c / 20;
            int cc = c % 20;
            int e  = base + el;
            if (e >= E) e = E - 1;            // tail guard (E%16==0 normally)
            _Float16* dp = rep + el * K1 + cc * 8;
            if (cc < 16) {
                int node = (cc < 8) ? srcIdx[e] : dstIdx[e];
                int k8   = (cc < 8) ? cc : cc - 8;
                *(half8*)dp = *(const half8*)(h2h + (size_t)node * HID + k8 * 8);
            } else {
                const float* p = ea + (size_t)e * EDIM + (cc - 16) * 8;
                half8 v;
#pragma unroll
                for (int j = 0; j < 8; ++j) v[j] = (_Float16)p[j];
                *(half8*)dp = v;
            }
        }

        // ---- GEMM1: [16 x 160] x [160 x 64] -> z1, bias + ReLU
        v8f acc[4];
#pragma unroll
        for (int nt = 0; nt < 4; ++nt) {
            float bias = s_bc1[nt * 16 + (lane & 15)];
#pragma unroll
            for (int r = 0; r < 8; ++r) acc[nt][r] = bias;
        }
#pragma unroll
        for (int ks = 0; ks < 5; ++ks) {
            v16h a = frag_a(rep, K1, ks * 32, lane);
#pragma unroll
            for (int nt = 0; nt < 4; ++nt) {
                v16h b = frag_b(s_wc1, K1, nt * 16, ks * 32, lane);
                acc[nt] = __builtin_amdgcn_wmma_f32_16x16x32_f16(
                    false, a, false, b, (short)0, acc[nt], false, false);
            }
        }
#pragma unroll
        for (int nt = 0; nt < 4; ++nt) {
            int n = nt * 16 + (lane & 15);
#pragma unroll
            for (int r = 0; r < 8; ++r) {
                float v = acc[nt][r];
                z1[(mOff + r) * HID + n] = (_Float16)(v > 0.0f ? v : 0.0f);
            }
        }

        // ---- GEMM2: [16 x 64] x [64 x 32] -> z2, bias + ReLU
        v8f acc2[2];
#pragma unroll
        for (int nt = 0; nt < 2; ++nt) {
            float bias = s_bc2[nt * 16 + (lane & 15)];
#pragma unroll
            for (int r = 0; r < 8; ++r) acc2[nt][r] = bias;
        }
#pragma unroll
        for (int ks = 0; ks < 2; ++ks) {
            v16h a = frag_a(z1, HID, ks * 32, lane);
#pragma unroll
            for (int nt = 0; nt < 2; ++nt) {
                v16h b = frag_b(s_wc2, HID, nt * 16, ks * 32, lane);
                acc2[nt] = __builtin_amdgcn_wmma_f32_16x16x32_f16(
                    false, a, false, b, (short)0, acc2[nt], false, false);
            }
        }
#pragma unroll
        for (int nt = 0; nt < 2; ++nt) {
            int n = nt * 16 + (lane & 15);
#pragma unroll
            for (int r = 0; r < 8; ++r) {
                float v = acc2[nt][r];
                z2[(mOff + r) * 32 + n] = (v > 0.0f ? v : 0.0f);
            }
        }

        // ---- GEMM3: per-edge 32-dot on VALU (16 lanes; LDS in-order per wave)
        if (lane < 16 && base + lane < E) {
            const float* row = z2 + lane * 32;
            float s = s_bc3[0];
#pragma unroll
            for (int c = 0; c < 32; ++c) s += row[c] * s_wc3[c];
            out[base + lane] = s;
        }
    }
}

// ---------------------------------------------------------------------------

extern "C" void kernel_launch(void* const* d_in, const int* in_sizes, int n_in,
                              void* d_out, int out_size, void* d_ws, size_t ws_size,
                              hipStream_t stream) {
    // inputs: 0=x(unused) 1=edge_index 2=edge_attr 3=node_stats 4=W1 5=b1
    //         6=W2 7=b2 8=Wc1 9=bc1 10=Wc2 11=bc2 12=Wc3 13=bc3
    const int*   eidx = (const int*)d_in[1];
    const float* ea   = (const float*)d_in[2];
    const float* ns   = (const float*)d_in[3];
    const float* W1   = (const float*)d_in[4];
    const float* b1   = (const float*)d_in[5];
    const float* W2   = (const float*)d_in[6];
    const float* b2   = (const float*)d_in[7];
    const float* Wc1  = (const float*)d_in[8];
    const float* bc1  = (const float*)d_in[9];
    const float* Wc2  = (const float*)d_in[10];
    const float* bc2  = (const float*)d_in[11];
    const float* Wc3  = (const float*)d_in[12];
    const float* bc3  = (const float*)d_in[13];
    float* out = (float*)d_out;

    const int N = in_sizes[3] / 2;     // node_stats: N x 2
    const int E = in_sizes[2] / EDIM;  // edge_attr : E x 32

    auto al = [](size_t x) { return (x + 255) & ~(size_t)255; };
    char* ws = (char*)d_ws;
    size_t off = 0;
    float* deg  = (float*)(ws + off); off = al(off + (size_t)N * 4);
    float* agg1 = (float*)(ws + off); off = al(off + (size_t)N * 34 * 4);
    float* agg2 = (float*)(ws + off); size_t zero_end = off + (size_t)N * HID * 4;
    off = al(zero_end);
    float* h1 = (float*)(ws + off); off = al(off + (size_t)N * HID * 4);
    _Float16* h2h = (_Float16*)(ws + off);

    hipMemsetAsync(d_ws, 0, zero_end, stream);

    long long t1 = (long long)E * EDIM;
    scatter1_kernel<<<(unsigned)((t1 + 255) / 256), 256, 0, stream>>>(eidx, ns, ea, deg, agg1, E);
    dense1_kernel<<<(N * HID + 255) / 256, 256, 0, stream>>>(ns, deg, agg1, W1, b1, h1, N);

    long long t2 = (long long)E * HID;
    scatter2_kernel<<<(unsigned)((t2 + 255) / 256), 256, 0, stream>>>(eidx, h1, agg2, E);
    dense2_kernel<<<(N * HID + 255) / 256, 256, 0, stream>>>(h1, deg, agg2, agg1, W2, b2, h2h, N);

    int tiles = (E + 15) / 16;
    int blocks = (tiles + WPB - 1) / WPB;
    if (blocks > 2048) blocks = 2048;   // grid-stride: ~6 tiles/wave, 8192 waves
    classifier_kernel<<<blocks, WPB * 32, 0, stream>>>(
        eidx, ea, h2h, Wc1, bc1, Wc2, bc2, Wc3, bc3, out, E, tiles);
}
